// GatedMultiheadAttention_82154134438460
// MI455X (gfx1250) — compile-verified, matrix operand order checked by reading, not run on hardware
//
#include <hip/hip_runtime.h>

#define B_   4
#define T_   2048
#define DM_  1024
#define H_   16
#define D_   64
#define NTOK (B_ * T_)   // 8192 tokens

typedef __bf16 bf16;
typedef __attribute__((ext_vector_type(16))) __bf16 v16bf;
typedef __attribute__((ext_vector_type(8)))  __bf16 v8bf;
typedef __attribute__((ext_vector_type(8)))  float  v8f;

union Frag16 { v16bf v; v8bf h[2]; };

#if __has_builtin(__builtin_amdgcn_global_load_async_to_lds_b128) && \
    __has_builtin(__builtin_amdgcn_s_wait_asynccnt)
#define USE_ASYNC_COPY 1
typedef int v4i_vs __attribute__((vector_size(16)));          // matches builtin pointee
typedef __attribute__((address_space(1))) v4i_vs as1_v4i;     // global
typedef __attribute__((address_space(3))) v4i_vs as3_v4i;     // LDS
#else
#define USE_ASYNC_COPY 0
#endif

// Load a 16-bit A/B WMMA fragment (16x32 / 32x16) where the lane's 16 values are
// two contiguous 8-element runs in memory:
//   lanes 0-15 : K = k0+0..7   and k0+16..23
//   lanes 16-31: K = k0+8..15  and k0+24..31
// "row" is the M index (A, lane%16) or N index (B, from an N-major matrix).
__device__ __forceinline__ v16bf load_frag(const bf16* base, int row, int ld,
                                           int k0, int half) {
  Frag16 f;
  const bf16* p = base + (size_t)row * ld + k0 + half * 8;
  f.h[0] = *(const v8bf*)(p);
  f.h[1] = *(const v8bf*)(p + 16);
  return f.v;
}

__device__ __forceinline__ v8f wmma_bf16(v16bf a, v16bf b, v8f c) {
  // 8 args: (neg_a, A, neg_b, B, c_mod, C, reuse_a, reuse_b)
  return __builtin_amdgcn_wmma_f32_16x16x32_bf16(false, a, false, b, (short)0, c,
                                                 false, false);
}

// ---------------------------------------------------------------- stage 1: converts

__global__ void k_f32_to_bf16(const float* __restrict__ src, bf16* __restrict__ dst, int n) {
  int i = blockIdx.x * blockDim.x + threadIdx.x;
  int stride = gridDim.x * blockDim.x;
  for (; i < n; i += stride) dst[i] = (bf16)src[i];
}

// WT[n*rows + k] = W[k*cols + n]  (W is rows x cols = K x N, row-major)
__global__ void k_transpose_w(const float* __restrict__ W, bf16* __restrict__ WT,
                              int rows, int cols) {
  int i = blockIdx.x * blockDim.x + threadIdx.x;
  int total = rows * cols;
  int stride = gridDim.x * blockDim.x;
  for (; i < total; i += stride) {
    int k = i / cols, n = i % cols;
    WT[(size_t)n * rows + k] = (bf16)W[i];
  }
}

// gwT[h][e][k] = gw[h][k][e]
__global__ void k_transpose_gw(const float* __restrict__ gw, bf16* __restrict__ gwT) {
  int i = blockIdx.x * blockDim.x + threadIdx.x;
  int total = H_ * D_ * D_;
  int stride = gridDim.x * blockDim.x;
  for (; i < total; i += stride) {
    int h = i / (D_ * D_);
    int r = i % (D_ * D_);
    int k = r / D_, e = r % D_;
    gwT[(size_t)h * D_ * D_ + e * D_ + k] = (bf16)gw[i];
  }
}

// ---------------------------------------------------------------- big GEMMs
// X: (NTOK, DM) bf16 row-major.  WT: (N=DM, K=DM) bf16 N-major.  bias: fp32.
// mode 0: bf16 out[(b,h,t,d)]   (Q, K)
// mode 1: bf16 out[(b,h,d,t)]   (V stored transposed for contiguous PV B-frags)
// mode 2: fp32 out[(tok, n)]    (final output projection)
//
// Block = 8 waves = 128 (M) x 64 (N) tile. The 64x64 bf16 B tile for the
// current k-step is shared by all 8 waves, staged in LDS with a double buffer.
// With async-to-LDS support the staging is GLOBAL_LOAD_ASYNC_TO_LDS_B128
// (ASYNCcnt), issuing tile k+1 before s_wait_asynccnt(2) for tile k.
__device__ __forceinline__ void stage_b_tile(const bf16* __restrict__ gsrc,
                                             bf16* lds_dst, int tid) {
  // 64 rows x 64 bf16 (128B/row); 4 threads per row, 16 elems (32B) per thread.
  int r = tid >> 2, kc = (tid & 3) * 16;
  const bf16* g = gsrc + (size_t)r * DM_ + kc;
  bf16* l = lds_dst + r * 64 + kc;
#if USE_ASYNC_COPY
  __builtin_amdgcn_global_load_async_to_lds_b128((as1_v4i*)(void*)g,
                                                 (as3_v4i*)(void*)l, 0, 0);
  __builtin_amdgcn_global_load_async_to_lds_b128((as1_v4i*)(void*)(g + 8),
                                                 (as3_v4i*)(void*)(l + 8), 0, 0);
#else
  *(v8bf*)l       = *(const v8bf*)g;
  *(v8bf*)(l + 8) = *(const v8bf*)(g + 8);
#endif
}

__global__ __launch_bounds__(256) void k_gemm(const bf16* __restrict__ X,
                                              const bf16* __restrict__ WT,
                                              const float* __restrict__ bias,
                                              bf16* __restrict__ outb,
                                              float* __restrict__ outf, int mode) {
  __shared__ __align__(16) bf16 ldsB[2][64 * 64];   // 16 KB double buffer

  int tid  = threadIdx.x;
  int wave = tid >> 5;
  int lane = tid & 31;
  int half = lane >> 4;
  int ln   = lane & 15;
  int m0 = blockIdx.x * 128 + wave * 16;   // token rows
  int n0 = blockIdx.y * 64;                // output features

  const bf16* wbase = WT + (size_t)n0 * DM_;
  const int nTiles = DM_ / 64;             // 16 k-steps

  stage_b_tile(wbase, ldsB[0], tid);       // prefetch tile 0

  v8f acc[4] = {};
  for (int kt = 0; kt < nTiles; ++kt) {
    if (kt + 1 < nTiles) {
      stage_b_tile(wbase + (kt + 1) * 64, ldsB[(kt + 1) & 1], tid);
#if USE_ASYNC_COPY
      __builtin_amdgcn_s_wait_asynccnt(2);   // tile kt complete, kt+1 in flight
#endif
    } else {
#if USE_ASYNC_COPY
      __builtin_amdgcn_s_wait_asynccnt(0);
#endif
    }
    __syncthreads();

    int k0 = kt * 64;
    const bf16* lb = ldsB[kt & 1];
    v16bf a0 = load_frag(X, m0 + ln, DM_, k0, half);
    v16bf a1 = load_frag(X, m0 + ln, DM_, k0 + 32, half);
#pragma unroll
    for (int c = 0; c < 4; ++c) {
      v16bf b0 = load_frag(lb, c * 16 + ln, 64, 0, half);
      v16bf b1 = load_frag(lb, c * 16 + ln, 64, 32, half);
      acc[c] = wmma_bf16(a0, b0, acc[c]);
      acc[c] = wmma_bf16(a1, b1, acc[c]);
    }
    __syncthreads();                       // protect buffer before next overwrite
  }

#pragma unroll
  for (int c = 0; c < 4; ++c) {
    int n = n0 + c * 16 + ln;
    float bv = bias[n];
    int hh = n / D_, dd = n % D_;
#pragma unroll
    for (int j = 0; j < 8; ++j) {
      int m   = j + half * 8;
      int tok = m0 + m;
      float val = acc[c][j] + bv;
      if (mode == 2) {
        outf[(size_t)tok * DM_ + n] = val;
      } else {
        int bb = tok / T_, tt = tok % T_;
        size_t addr;
        if (mode == 0) addr = (((size_t)(bb * H_ + hh)) * T_ + tt) * D_ + dd;
        else           addr = (((size_t)(bb * H_ + hh)) * D_ + dd) * T_ + tt;
        outb[addr] = (bf16)val;
      }
    }
  }
}

// ---------------------------------------------------------------- flash attention + gating
// One wave (32 threads) per (b*h, 16-query tile).
__global__ __launch_bounds__(32) void k_attn_gate(const bf16* __restrict__ Qh,
                                                  const bf16* __restrict__ Kh,
                                                  const bf16* __restrict__ Vt,
                                                  const bf16* __restrict__ gwT,
                                                  const float* __restrict__ gate_b,
                                                  bf16* __restrict__ Yg) {
  __shared__ __align__(16) bf16 ldsP[16 * 32];  // P tile staging (C-layout -> A-layout)
  __shared__ __align__(16) bf16 ldsY[16 * 64];  // Y tile staging for gating GEMM

  int bh = blockIdx.x;                 // 0..B*H-1
  int q0 = blockIdx.y * 16;            // query tile start
  int bb = bh / H_;
  int hh = bh % H_;
  int lane = threadIdx.x;
  int half = lane >> 4;
  int ln   = lane & 15;

  const bf16* Qbase = Qh + (size_t)bh * T_ * D_;
  const bf16* Kbase = Kh + (size_t)bh * T_ * D_;
  const bf16* Vbase = Vt + (size_t)bh * D_ * T_;

  // Q A-fragments, K-dim 0..31 and 32..63
  v16bf qa0 = load_frag(Qbase, q0 + ln, D_, 0,  half);
  v16bf qa1 = load_frag(Qbase, q0 + ln, D_, 32, half);

  v8f O[4] = {};                       // 16 x 64 f32 accumulator
  float rm[8], rs[8];                  // per-row running max / sum (rows j + 8*half)
#pragma unroll
  for (int j = 0; j < 8; ++j) { rm[j] = -1e30f; rs[j] = 0.0f; }
  const float scale = 0.125f;          // 1/sqrt(64)

  for (int kk = 0; kk < T_; kk += 32) {
    // ---- scores: two 16-key tiles ----
    v8f s0 = {}, s1 = {};
    {
      v16bf b0 = load_frag(Kbase, kk + ln,      D_, 0,  half);
      v16bf b1 = load_frag(Kbase, kk + ln,      D_, 32, half);
      s0 = wmma_bf16(qa0, b0, s0);
      s0 = wmma_bf16(qa1, b1, s0);
      v16bf b2 = load_frag(Kbase, kk + 16 + ln, D_, 0,  half);
      v16bf b3 = load_frag(Kbase, kk + 16 + ln, D_, 32, half);
      s1 = wmma_bf16(qa0, b2, s1);
      s1 = wmma_bf16(qa1, b3, s1);
    }
    // ---- prefetch V fragments; softmax VALU below hides their latency ----
    v16bf vb0 = load_frag(Vbase, 0 * 16 + ln, T_, kk, half);
    v16bf vb1 = load_frag(Vbase, 1 * 16 + ln, T_, kk, half);
    v16bf vb2 = load_frag(Vbase, 2 * 16 + ln, T_, kk, half);
    v16bf vb3 = load_frag(Vbase, 3 * 16 + ln, T_, kk, half);

    // ---- online softmax (row-wise over the 16 lanes of each half) ----
#pragma unroll
    for (int j = 0; j < 8; ++j) {
      float v0 = s0[j] * scale, v1 = s1[j] * scale;
      float mx = fmaxf(v0, v1);
#pragma unroll
      for (int msk = 1; msk < 16; msk <<= 1) mx = fmaxf(mx, __shfl_xor(mx, msk, 32));
      float nm = fmaxf(rm[j], mx);
      float cf = __expf(rm[j] - nm);
      float p0 = __expf(v0 - nm);
      float p1 = __expf(v1 - nm);
      float ps = p0 + p1;
#pragma unroll
      for (int msk = 1; msk < 16; msk <<= 1) ps += __shfl_xor(ps, msk, 32);
      rs[j] = rs[j] * cf + ps;
      rm[j] = nm;
      s0[j] = p0;
      s1[j] = p1;
#pragma unroll
      for (int c = 0; c < 4; ++c) O[c][j] *= cf;
    }
    // ---- P: C-layout -> A-layout via LDS (wave-private, in-order per wave) ----
#pragma unroll
    for (int j = 0; j < 8; ++j) {
      int m = j + half * 8;
      ldsP[m * 32 + ln]      = (bf16)s0[j];
      ldsP[m * 32 + 16 + ln] = (bf16)s1[j];
    }
    v16bf pa = load_frag(ldsP, ln, 32, 0, half);
    // ---- P @ V with prefetched fragments ----
    O[0] = wmma_bf16(pa, vb0, O[0]);
    O[1] = wmma_bf16(pa, vb1, O[1]);
    O[2] = wmma_bf16(pa, vb2, O[2]);
    O[3] = wmma_bf16(pa, vb3, O[3]);
  }

  // ---- normalize and stage Y for the gating GEMM ----
#pragma unroll
  for (int j = 0; j < 8; ++j) {
    float inv = 1.0f / rs[j];
    int m = j + half * 8;
#pragma unroll
    for (int c = 0; c < 4; ++c) {
      O[c][j] *= inv;
      ldsY[m * 64 + c * 16 + ln] = (bf16)O[c][j];
    }
  }
  v16bf ya0 = load_frag(ldsY, ln, 64, 0,  half);
  v16bf ya1 = load_frag(ldsY, ln, 64, 32, half);

  const bf16* gbase = gwT + (size_t)hh * D_ * D_;
#pragma unroll
  for (int c = 0; c < 4; ++c) {
    v8f g = {};
    v16bf gb0 = load_frag(gbase, c * 16 + ln, D_, 0,  half);
    v16bf gb1 = load_frag(gbase, c * 16 + ln, D_, 32, half);
    g = wmma_bf16(ya0, gb0, g);
    g = wmma_bf16(ya1, gb1, g);
    float bg = gate_b[hh * D_ + c * 16 + ln];
#pragma unroll
    for (int j = 0; j < 8; ++j) {
      float gv = g[j] + bg;
      float sg = 1.0f / (1.0f + __expf(-gv));
      float yv = O[c][j] * sg;
      int m   = j + half * 8;
      int tok = bb * T_ + q0 + m;                       // merged-head row-major out
      Yg[(size_t)tok * DM_ + hh * D_ + c * 16 + ln] = (bf16)yv;
    }
  }
}

// ---------------------------------------------------------------- host launcher

extern "C" void kernel_launch(void* const* d_in, const int* in_sizes, int n_in,
                              void* d_out, int out_size, void* d_ws, size_t ws_size,
                              hipStream_t stream) {
  (void)in_sizes; (void)n_in; (void)out_size; (void)ws_size;
  const float* q  = (const float*)d_in[0];
  const float* k  = (const float*)d_in[1];
  const float* v  = (const float*)d_in[2];
  const float* Wq = (const float*)d_in[3];
  const float* bq = (const float*)d_in[4];
  const float* Wk = (const float*)d_in[5];
  const float* bk = (const float*)d_in[6];
  const float* Wv = (const float*)d_in[7];
  const float* bv = (const float*)d_in[8];
  const float* Wo = (const float*)d_in[9];
  const float* bo = (const float*)d_in[10];
  const float* gw = (const float*)d_in[11];
  const float* gb = (const float*)d_in[12];
  float* out = (float*)d_out;

  char* ws = (char*)d_ws;
  size_t off = 0;
  auto alloc = [&](size_t bytes) {
    char* p = ws + off;
    off += (bytes + 255) & ~(size_t)255;
    return p;
  };
  const size_t actBytes = (size_t)NTOK * DM_ * sizeof(bf16);  // 16 MB
  const size_t wBytes   = (size_t)DM_ * DM_ * sizeof(bf16);   //  2 MB
  bf16* Xq  = (bf16*)alloc(actBytes);
  bf16* Xk  = (bf16*)alloc(actBytes);
  bf16* Xv  = (bf16*)alloc(actBytes);
  bf16* WqT = (bf16*)alloc(wBytes);
  bf16* WkT = (bf16*)alloc(wBytes);
  bf16* WvT = (bf16*)alloc(wBytes);
  bf16* WoT = (bf16*)alloc(wBytes);
  bf16* gwT = (bf16*)alloc((size_t)H_ * D_ * D_ * sizeof(bf16));
  bf16* Qh  = (bf16*)alloc(actBytes);
  bf16* Kh  = (bf16*)alloc(actBytes);
  bf16* VtW = (bf16*)alloc(actBytes);
  bf16* Yg  = (bf16*)alloc(actBytes);

  const int nAct = NTOK * DM_;
  k_f32_to_bf16<<<4096, 256, 0, stream>>>(q, Xq, nAct);
  k_f32_to_bf16<<<4096, 256, 0, stream>>>(k, Xk, nAct);
  k_f32_to_bf16<<<4096, 256, 0, stream>>>(v, Xv, nAct);
  k_transpose_w<<<2048, 256, 0, stream>>>(Wq, WqT, DM_, DM_);
  k_transpose_w<<<2048, 256, 0, stream>>>(Wk, WkT, DM_, DM_);
  k_transpose_w<<<2048, 256, 0, stream>>>(Wv, WvT, DM_, DM_);
  k_transpose_w<<<2048, 256, 0, stream>>>(Wo, WoT, DM_, DM_);
  k_transpose_gw<<<128, 256, 0, stream>>>(gw, gwT);

  dim3 gemmGrid(NTOK / 128, DM_ / 64);   // 64 x 16
  k_gemm<<<gemmGrid, 256, 0, stream>>>(Xq, WqT, bq, Qh, nullptr, 0);
  k_gemm<<<gemmGrid, 256, 0, stream>>>(Xk, WkT, bk, Kh, nullptr, 0);
  k_gemm<<<gemmGrid, 256, 0, stream>>>(Xv, WvT, bv, VtW, nullptr, 1);

  dim3 attnGrid(B_ * H_, T_ / 16);       // 64 x 128
  k_attn_gate<<<attnGrid, 32, 0, stream>>>(Qh, Kh, VtW, gwT, gb, Yg);

  k_gemm<<<gemmGrid, 256, 0, stream>>>(Yg, WoT, bo, nullptr, out, 2);
}
